// BaseBackbone_19868518711489
// MI455X (gfx1250) — matrix operand order repaired and checked by reading, not compiled, not run on hardware
//
#include <hip/hip_runtime.h>
#include <hip/hip_bf16.h>
#include <math.h>

typedef __bf16 bf16;
typedef __attribute__((ext_vector_type(16))) __bf16 v16bf;
typedef __attribute__((ext_vector_type(8)))  __bf16 v8bf;
typedef __attribute__((ext_vector_type(8)))  float  v8f;

// ---------------- constants ----------------
#define Bd   16
#define CHd  16
#define Ed   384
#define NHd  6
#define HDd  64
#define MLPd 1536

// d_out layout (float elements)
static const size_t OUT0   = 0;                       // (16,320,384)
static const size_t Z_OFF  = 1966080;
static const size_t X_OFF  = 6160384;
static const size_t ZCG_OFF= 22937600;
static const size_t XCG_OFF= 27131904;
static const size_t OV_OFF = 43909120;
static const size_t OI_OFF = 43909376;

// ws layout (bytes)
static const size_t O_WQKV  = 0;
static const size_t O_WPROJ = 3538944;
static const size_t O_WMLP1 = 4718592;
static const size_t O_WMLP2 = 9437184;
static const size_t O_WPATCH= 14155776;
static const size_t O_CONV  = 14745600;
static const size_t O_TCAT  = 15540224;
static const size_t O_MSM   = 15619072;
static const size_t O_W     = 15697920;
static const size_t O_OIDX  = 15698944;
static const size_t O_TZ    = 15699968;
static const size_t O_TX    = 23564288;
static const size_t O_TC    = 55021568;
static const size_t O_TSUM  = 94343168;
static const size_t O_H     = 102207488;
static const size_t O_QKV   = 121868288;
static const size_t O_OB    = 180850688;
static const size_t O_U     = 200511488;

// ---------------- WMMA fragment helpers ----------------
__device__ __forceinline__ v8f wmma_bf16(v16bf a, v16bf b, v8f c) {
  return __builtin_amdgcn_wmma_f32_16x16x32_bf16(false, a, false, b, (short)0, c, false, false);
}

// A fragment (16x32, row = lane%16). element e<8 -> K = base+e ; e>=8 -> K = base+16+(e-8)
// caller passes p0 = &A[row*lda + k0 + (lane>=16 ? 8 : 0)]
__device__ __forceinline__ v16bf load_a_f32(const float* __restrict__ p0) {
  v16bf a;
#pragma unroll
  for (int e = 0; e < 8; ++e) a[e] = (bf16)p0[e];
#pragma unroll
  for (int e = 0; e < 8; ++e) a[8 + e] = (bf16)p0[16 + e];
  return a;
}
__device__ __forceinline__ v16bf load_a_bf16(const bf16* __restrict__ p0) {
  v16bf a;
  v8bf lo = *(const v8bf*)p0;
  v8bf hi = *(const v8bf*)(p0 + 16);
#pragma unroll
  for (int e = 0; e < 8; ++e) { a[e] = lo[e]; a[8 + e] = hi[e]; }
  return a;
}
// B fragment from Bt[N][K] (row-major B^T). col = lane%16, e<8 -> K = base+e ; e>=8 -> K = base+16+(e-8)
// caller passes p0 = &Bt[(n)*ldK + k0 + (lane>=16 ? 8 : 0)]
__device__ __forceinline__ v16bf load_b(const bf16* __restrict__ p0) {
  v16bf b;
  v8bf lo = *(const v8bf*)p0;
  v8bf hi = *(const v8bf*)(p0 + 16);
#pragma unroll
  for (int e = 0; e < 8; ++e) { b[e] = lo[e]; b[8 + e] = hi[e]; }
  return b;
}

__device__ __forceinline__ float gelu_tanh(float x) {
  return 0.5f * x * (1.f + tanhf(0.7978845608028654f * (x + 0.044715f * x * x * x)));
}

// ---------------- generic bf16 WMMA GEMM with async-to-LDS B staging ----------------
// C[M,N] = epi(A @ Bt^T + bias [+ Res]) ; Bt is [N][K] bf16, K multiple of 128.
// block = 128 threads (4 waves), wave tile 16x64, block tile 64x64.
// B tile (64 x 128 halves) staged in LDS via GLOBAL_LOAD_ASYNC_TO_LDS_B128, double buffered.
template <int AMODE, int BIAS, int GELU_, int RES, int OUTBF>
__global__ __launch_bounds__(128) void gemm_k(const void* __restrict__ Av, int lda,
                                              const bf16* __restrict__ Bt,
                                              const float* __restrict__ bias,
                                              const float* __restrict__ Res, int ldr,
                                              void* __restrict__ Cv, int ldc, int K) {
  const int tid = threadIdx.x;
  const int lane = tid & 31, wid = tid >> 5;
  const int hi = lane >> 4, lm = lane & 15;
  const int m0 = blockIdx.y * 64 + wid * 16;
  const int n0 = blockIdx.x * 64;

  __shared__ __align__(16) bf16 ldsB[2][64 * 128];  // 32KB double buffer

  const unsigned lds0 = (unsigned)(uintptr_t)&ldsB[0][0];
  const unsigned lds1 = (unsigned)(uintptr_t)&ldsB[1][0];

  v8f acc[4] = {};
  const size_t arow = (size_t)(m0 + lm);
  const int S = K >> 7;  // slabs of 128

  // issue one 64x128-half slab of Bt into LDS buffer (8 async b128 ops per thread)
  auto issue = [&](int s, unsigned lbase) {
#pragma unroll
    for (int j = 0; j < 8; ++j) {
      const int ch = tid + j * 128;                       // 16-byte chunk id (1024 total)
      const unsigned ldsoff = lbase + (unsigned)(ch * 16);
      const unsigned long long gp =
          (unsigned long long)(Bt + (size_t)(n0 + (ch >> 4)) * K + (s << 7) + (ch & 15) * 8);
      asm volatile("global_load_async_to_lds_b128 %0, %1, off" ::"v"(ldsoff), "v"(gp)
                   : "memory");
    }
  };

  issue(0, lds0);
  for (int s = 0; s < S; ++s) {
    if (s + 1 < S) {
      issue(s + 1, (s & 1) ? lds0 : lds1);
      asm volatile("s_wait_asynccnt 0x8" ::: "memory");  // slab s retired (in-order), s+1 in flight
    } else {
      asm volatile("s_wait_asynccnt 0x0" ::: "memory");
    }
    __syncthreads();  // slab s visible to all waves
    const bf16* bb = &ldsB[s & 1][0];
    const int kg = s << 7;
#pragma unroll
    for (int kk = 0; kk < 4; ++kk) {
      const int k0 = kg + kk * 32;
      v16bf a;
      if (AMODE == 0) a = load_a_f32((const float*)Av + arow * lda + k0 + hi * 8);
      else            a = load_a_bf16((const bf16*)Av + arow * lda + k0 + hi * 8);
#pragma unroll
      for (int t = 0; t < 4; ++t) {
        const bf16* bp = bb + (t * 16 + lm) * 128 + kk * 32 + hi * 8;
        acc[t] = wmma_bf16(a, load_b(bp), acc[t]);
      }
    }
    __syncthreads();  // all waves done reading this buffer before it is re-issued
  }

#pragma unroll
  for (int t = 0; t < 4; ++t) {
    const int col = n0 + t * 16 + lm;
#pragma unroll
    for (int i = 0; i < 8; ++i) {
      const size_t row = (size_t)(m0 + i + hi * 8);
      float v = acc[t][i];
      if (BIAS)  v += bias[col];
      if (GELU_) v = gelu_tanh(v);
      if (RES)   v += Res[row * ldr + col];
      if (OUTBF) ((bf16*)Cv)[row * ldc + col] = (bf16)v;
      else       ((float*)Cv)[row * ldc + col] = v;
    }
  }
}

// ---------------- patch-embed GEMM (fused gather + normalize) ----------------
// M = 80*tok rows, N = 384, K = 768 (cg,ky,kx). img is (16,16,H,H) f32.
__global__ __launch_bounds__(128) void patch_k(const float* __restrict__ img, int Himg,
                                               int tok, int side,
                                               const int* __restrict__ oidx,
                                               const bf16* __restrict__ Bt,
                                               const float* __restrict__ bias,
                                               float* __restrict__ C) {
  const int lane = threadIdx.x & 31, wid = threadIdx.x >> 5;
  const int hi = lane >> 4, lm = lane & 15;
  const int m0 = blockIdx.y * 64 + wid * 16;
  const int n0 = blockIdx.x * 64;
  const int row = m0 + lm;
  const int g = row / (Bd * tok);
  const int r = row % (Bd * tok);
  const int b = r / tok;
  const int p = r % tok;
  const int py = p / side, px = p % side;
  const float* ib = img + (size_t)b * CHd * Himg * Himg;
  v8f acc[4] = {};
  for (int k0 = 0; k0 < 768; k0 += 32) {
    v16bf a;
#pragma unroll
    for (int half = 0; half < 2; ++half) {
      const int kb = k0 + hi * 8 + half * 16;
      const int cg = kb >> 8, rem = kb & 255, ky = rem >> 4, kx = rem & 15;
      const int csrc = oidx[b * 16 + 3 * g + cg];
      const float mu = (cg == 0) ? 0.485f : (cg == 1) ? 0.456f : 0.406f;
      const float is = (cg == 0) ? (1.f / 0.229f) : (cg == 1) ? (1.f / 0.224f) : (1.f / 0.225f);
      const float* pp = ib + (size_t)csrc * Himg * Himg + (size_t)(py * 16 + ky) * Himg + (px * 16 + kx);
#pragma unroll
      for (int e = 0; e < 8; ++e) a[half * 8 + e] = (bf16)((pp[e] - mu) * is);
    }
#pragma unroll
    for (int t = 0; t < 4; ++t) {
      const bf16* bp = Bt + (size_t)(n0 + t * 16 + lm) * 768 + k0 + hi * 8;
      acc[t] = wmma_bf16(a, load_b(bp), acc[t]);
    }
  }
#pragma unroll
  for (int t = 0; t < 4; ++t) {
    const int col = n0 + t * 16 + lm;
#pragma unroll
    for (int i = 0; i < 8; ++i) {
      const size_t orow = (size_t)(m0 + i + hi * 8);
      C[orow * Ed + col] = acc[t][i] + bias[col];
    }
  }
}

// ---------------- fused flash attention ----------------
// one workgroup (4 waves) per (batch-of-80, head). qkv bf16 [rows][1152]; ob bf16 [rows][384].
template <int NT>
__global__ __launch_bounds__(128) void attn_k(const bf16* __restrict__ qkv, bf16* __restrict__ ob) {
  constexpr int NQB = NT / 16;
  constexpr int QPW = (NQB + 3) / 4;
  const int bh = blockIdx.x;
  const int bi = bh / NHd, h = bh % NHd;
  const size_t base = (size_t)bi * NT;
  const int tid = threadIdx.x;
  const int lane = tid & 31, wid = tid >> 5;
  const int hi = lane >> 4, lm = lane & 15;

  __shared__ __align__(16) bf16 kb_lds[128 * 64];
  __shared__ __align__(16) bf16 vT_lds[64 * 136];
  __shared__ __align__(16) bf16 p_lds[4 * 16 * 128];

  v8f oacc[QPW][4] = {};
  float mst[QPW][8], lst[QPW][8];
#pragma unroll
  for (int q = 0; q < QPW; ++q)
#pragma unroll
    for (int i = 0; i < 8; ++i) { mst[q][i] = -1e30f; lst[q][i] = 0.f; }

#pragma unroll
  for (int kv0 = 0; kv0 < NT; kv0 += 128) {
    const int C = (NT - kv0) < 128 ? (NT - kv0) : 128;
    for (int idx = tid; idx < C * 64; idx += 128) {
      const int kv = idx >> 6, d = idx & 63;
      const bf16* src = qkv + (base + kv0 + kv) * 1152 + h * 64 + d;
      kb_lds[kv * 64 + d] = src[384];   // K
      vT_lds[d * 136 + kv] = src[768];  // V^T
    }
    __syncthreads();
#pragma unroll
    for (int qi = 0; qi < QPW; ++qi) {
      const int qb = wid + qi * 4;
      if (qb < NQB) {
        const bf16* qp = qkv + (base + qb * 16 + lm) * 1152 + h * 64;
        const v16bf qf0 = load_a_bf16(qp + 0  + hi * 8);
        const v16bf qf1 = load_a_bf16(qp + 32 + hi * 8);
        const int NTile = C >> 4;
        v8f st[8];
        float cmax[8];
#pragma unroll
        for (int i = 0; i < 8; ++i) cmax[i] = -1e30f;
#pragma unroll
        for (int kt = 0; kt < 8; ++kt) if (kt < NTile) {
          v8f s = {};
          const bf16* b0 = kb_lds + (kt * 16 + lm) * 64 + hi * 8;
          s = wmma_bf16(qf0, load_b(b0), s);
          s = wmma_bf16(qf1, load_b(b0 + 32), s);
#pragma unroll
          for (int i = 0; i < 8; ++i) {
            float v = s[i] * 0.125f;  // 1/sqrt(64)
            st[kt][i] = v;
#pragma unroll
            for (int m = 8; m; m >>= 1) v = fmaxf(v, __shfl_xor(v, m, 32));
            cmax[i] = fmaxf(cmax[i], v);
          }
        }
        float alpha[8], rs[8];
#pragma unroll
        for (int i = 0; i < 8; ++i) {
          const float nm = fmaxf(mst[qi][i], cmax[i]);
          alpha[i] = __expf(mst[qi][i] - nm);
          mst[qi][i] = nm;
          rs[i] = 0.f;
        }
#pragma unroll
        for (int kt = 0; kt < 8; ++kt) if (kt < NTile) {
#pragma unroll
          for (int i = 0; i < 8; ++i) {
            const float e = __expf(st[kt][i] - mst[qi][i]);
            rs[i] += e;
            p_lds[(wid * 16 + i + hi * 8) * 128 + kt * 16 + lm] = (bf16)e;
          }
        }
#pragma unroll
        for (int i = 0; i < 8; ++i) {
          float r = rs[i];
#pragma unroll
          for (int m = 8; m; m >>= 1) r += __shfl_xor(r, m, 32);
          lst[qi][i] = lst[qi][i] * alpha[i] + r;
#pragma unroll
          for (int t = 0; t < 4; ++t) oacc[qi][t][i] *= alpha[i];
        }
#pragma unroll
        for (int kk = 0; kk < 4; ++kk) if (kk < (C >> 5)) {
          const v16bf pf = load_a_bf16(p_lds + (wid * 16 + lm) * 128 + kk * 32 + hi * 8);
#pragma unroll
          for (int t = 0; t < 4; ++t) {
            const bf16* vb = vT_lds + (t * 16 + lm) * 136 + kk * 32 + hi * 8;
            oacc[qi][t] = wmma_bf16(pf, load_b(vb), oacc[qi][t]);
          }
        }
      }
    }
    __syncthreads();
  }
#pragma unroll
  for (int qi = 0; qi < QPW; ++qi) {
    const int qb = wid + qi * 4;
    if (qb < NQB) {
#pragma unroll
      for (int i = 0; i < 8; ++i) {
        const float inv = 1.f / lst[qi][i];
        const size_t row = base + qb * 16 + i + hi * 8;
#pragma unroll
        for (int t = 0; t < 4; ++t)
          ob[row * Ed + h * 64 + t * 16 + lm] = (bf16)(oacc[qi][t][i] * inv);
      }
    }
  }
}

// ---------------- LayerNorm (wave per row) ----------------
template <int OUTBF>
__global__ void ln_k(const float* __restrict__ X, const float* __restrict__ s,
                     const float* __restrict__ b, void* __restrict__ Y) {
  const int row = blockIdx.x * (blockDim.x >> 5) + (threadIdx.x >> 5);
  const int lane = threadIdx.x & 31;
  const float* xr = X + (size_t)row * Ed;
  float sum = 0.f, sq = 0.f;
  for (int e = lane; e < Ed; e += 32) { const float v = xr[e]; sum += v; sq += v * v; }
#pragma unroll
  for (int m = 16; m; m >>= 1) { sum += __shfl_xor(sum, m, 32); sq += __shfl_xor(sq, m, 32); }
  const float mu = sum * (1.f / Ed);
  const float var = sq * (1.f / Ed) - mu * mu;
  const float r = rsqrtf(var + 1e-6f);
  for (int e = lane; e < Ed; e += 32) {
    const float v = (xr[e] - mu) * r * s[e] + b[e];
    if (OUTBF) ((bf16*)Y)[(size_t)row * Ed + e] = (bf16)v;
    else       ((float*)Y)[(size_t)row * Ed + e] = v;
  }
}

// ---------------- small kernels ----------------
__global__ void conv_kern(const float* __restrict__ X, const float* __restrict__ Wc,
                          float* __restrict__ Y, int KS, int OH) {
  const int idx = blockIdx.x * blockDim.x + threadIdx.x;
  const int total = Bd * CHd * OH * OH;
  if (idx >= total) return;
  const int ox = idx % OH;
  int t = idx / OH;
  const int oy = t % OH; t /= OH;
  const int co = t % CHd; const int b = t / CHd;
  const float* xb = X + (size_t)b * CHd * 256 * 256;
  const float* wr = Wc + (size_t)co * CHd * KS * KS;
  float acc = 0.f;
  for (int ci = 0; ci < CHd; ++ci) {
    const float* xp = xb + (size_t)ci * 256 * 256 + (size_t)(oy * KS) * 256 + ox * KS;
    const float* wp = wr + ci * KS * KS;
    for (int ky = 0; ky < KS; ++ky)
      for (int kx = 0; kx < KS; ++kx) acc += xp[ky * 256 + kx] * wp[ky * KS + kx];
  }
  Y[idx] = acc;
}

__global__ void pool_kern(const float* __restrict__ Ysrc, float* __restrict__ tcat,
                          int OH, int PW, int loff) {
  const int idx = blockIdx.x * blockDim.x + threadIdx.x;
  const int total = Bd * CHd * PW * PW;
  if (idx >= total) return;
  const int lx = idx % PW;
  int t = idx / PW;
  const int ly = t % PW; t /= PW;
  const int c = t % CHd; const int b = t / CHd;
  float s = 0.f, mx = -1e30f;
  for (int dy = 0; dy < 3; ++dy)
    for (int dx = 0; dx < 3; ++dx) {
      const float v = Ysrc[((size_t)(b * CHd + c) * OH + ly * 3 + dy) * OH + lx * 3 + dx];
      s += v; mx = fmaxf(mx, v);
    }
  tcat[((size_t)b * 77 + loff + ly * PW + lx) * CHd + c] = s * (1.f / 9.f) + mx;
}

__global__ void msm_kern(const float* __restrict__ tcat, const float* __restrict__ Wmp,
                         float* __restrict__ msm) {
  const int idx = blockIdx.x * blockDim.x + threadIdx.x;
  if (idx >= Bd * 77 * CHd) return;
  const int c = idx % CHd;
  const int l = (idx / CHd) % 77;
  const int b = idx / (CHd * 77);
  float acc = 0.f;
  for (int d = 0; d < CHd; ++d) acc += tcat[((size_t)b * 77 + l) * CHd + d] * Wmp[d * CHd + c];
  msm[idx] = acc;
}

__global__ void cm_kern(const float* __restrict__ msm, float* __restrict__ wvec,
                        int* __restrict__ oidx_ws, float* __restrict__ ovals_out,
                        int* __restrict__ oidx_out) {
  const int b = blockIdx.x, tid = threadIdx.x;
  __shared__ float cm[16][16];
  __shared__ float rn[16];
  __shared__ float wl[16];
  const int c = tid >> 4, d = tid & 15;
  float acc = 0.f;
  for (int l = 0; l < 77; ++l)
    acc += msm[((size_t)b * 77 + l) * CHd + c] * msm[((size_t)b * 77 + l) * CHd + d];
  if (c == d) acc = 0.f;
  cm[c][d] = acc;
  __syncthreads();
  if (tid < 16) {
    float s = 0.f;
    for (int j = 0; j < 16; ++j) { const float v = cm[tid][j]; s += v * v; }
    rn[tid] = fmaxf(sqrtf(s), 1e-12f);
  }
  __syncthreads();
  float nv = cm[c][d] / rn[c];
  if (c == d) nv = 0.f;
  cm[c][d] = nv;
  __syncthreads();
  if (tid < 16) {
    float s = 0.f;
    for (int j = 0; j < 16; ++j) s += fabsf(cm[tid][j]);
    wl[tid] = s;
    wvec[b * 16 + tid] = s;
  }
  __syncthreads();
  if (tid < 16) {
    const float wc = wl[tid];
    int rank = 0;
    for (int j = 0; j < 16; ++j) rank += (wl[j] > wc) || (wl[j] == wc && j < tid);
    ovals_out[b * 16 + rank] = wc;
    oidx_out[b * 16 + rank] = tid;
    oidx_ws[b * 16 + rank] = tid;
  }
}

__global__ void cg_kern(const float* __restrict__ src, const float* __restrict__ wvec,
                        float* __restrict__ out_copy, float* __restrict__ out_cg, int HW) {
  const int idx = blockIdx.x * blockDim.x + threadIdx.x;
  if (idx >= Bd * CHd * HW) return;
  const int c = (idx / HW) & 15;
  const int b = idx / (HW * CHd);
  const float v = src[idx];
  out_copy[idx] = v;
  out_cg[idx] = v * wvec[b * 16 + c];
}

__global__ void cat_kern(const float* __restrict__ tz, const float* __restrict__ tx,
                         const float* __restrict__ pz, const float* __restrict__ px,
                         float* __restrict__ tc) {
  const int idx = blockIdx.x * blockDim.x + threadIdx.x;
  if (idx >= 80 * 320 * Ed) return;
  const int e = idx % Ed;
  const int n = (idx / Ed) % 320;
  const int bi = idx / (Ed * 320);
  float v;
  if (n < 64) v = tz[((size_t)bi * 64 + n) * Ed + e] + pz[n * Ed + e];
  else        v = tx[((size_t)bi * 256 + (n - 64)) * Ed + e] + px[(n - 64) * Ed + e];
  tc[idx] = v;
}

__global__ void sum5_kern(const float* __restrict__ tc, float* __restrict__ ts) {
  const int idx = blockIdx.x * blockDim.x + threadIdx.x;
  if (idx >= Bd * 320 * Ed) return;
  const int b = idx / (320 * Ed);
  const int r = idx % (320 * Ed);
  float acc = 0.f;
  for (int g = 0; g < 5; ++g) acc += tc[((size_t)(g * Bd + b)) * 320 * Ed + r];
  ts[idx] = acc;
}

__global__ void tcvt_kern(const float* __restrict__ Wsrc, bf16* __restrict__ Bt, int K, int N) {
  const int idx = blockIdx.x * blockDim.x + threadIdx.x;
  if (idx >= K * N) return;
  const int n = idx % N, k = idx / N;
  Bt[(size_t)n * K + k] = (bf16)Wsrc[idx];
}

__global__ void cvt_kern(const float* __restrict__ Wsrc, bf16* __restrict__ Bt, int n) {
  const int idx = blockIdx.x * blockDim.x + threadIdx.x;
  if (idx < n) Bt[idx] = (bf16)Wsrc[idx];
}

// ---------------- host orchestration ----------------
struct Ctx {
  bf16 *WqkvT, *WprojT, *Wmlp1T, *Wmlp2T;
  bf16 *hbf, *qkvbf, *obf, *ubf;
  const float *ln1_s, *ln1_b, *ln2_s, *ln2_b;
  hipStream_t st;
};

static void run_block(const Ctx& c, float* t, int M, int NT, int i) {
  ln_k<1><<<M / 8, 256, 0, c.st>>>(t, c.ln1_s + i * Ed, c.ln1_b + i * Ed, c.hbf);
  gemm_k<1, 0, 0, 0, 1><<<dim3(1152 / 64, M / 64), 128, 0, c.st>>>(
      c.hbf, Ed, c.WqkvT + (size_t)i * 1152 * Ed, nullptr, nullptr, 0, c.qkvbf, 1152, Ed);
  if (NT == 64)       attn_k<64><<<480, 128, 0, c.st>>>(c.qkvbf, c.obf);
  else if (NT == 256) attn_k<256><<<480, 128, 0, c.st>>>(c.qkvbf, c.obf);
  else                attn_k<320><<<480, 128, 0, c.st>>>(c.qkvbf, c.obf);
  gemm_k<1, 0, 0, 1, 0><<<dim3(Ed / 64, M / 64), 128, 0, c.st>>>(
      c.obf, Ed, c.WprojT + (size_t)i * Ed * Ed, nullptr, t, Ed, t, Ed, Ed);
  ln_k<1><<<M / 8, 256, 0, c.st>>>(t, c.ln2_s + i * Ed, c.ln2_b + i * Ed, c.hbf);
  gemm_k<1, 0, 1, 0, 1><<<dim3(MLPd / 64, M / 64), 128, 0, c.st>>>(
      c.hbf, Ed, c.Wmlp1T + (size_t)i * MLPd * Ed, nullptr, nullptr, 0, c.ubf, MLPd, Ed);
  gemm_k<1, 0, 0, 1, 0><<<dim3(Ed / 64, M / 64), 128, 0, c.st>>>(
      c.ubf, MLPd, c.Wmlp2T + (size_t)i * Ed * MLPd, nullptr, t, Ed, t, Ed, MLPd);
}

extern "C" void kernel_launch(void* const* d_in, const int* in_sizes, int n_in,
                              void* d_out, int out_size, void* d_ws, size_t ws_size,
                              hipStream_t stream) {
  const float* z      = (const float*)d_in[0];
  const float* x      = (const float*)d_in[1];
  const float* cs18_w = (const float*)d_in[2];
  const float* cs16_w = (const float*)d_in[3];
  const float* cs14_w = (const float*)d_in[4];
  const float* csMP_w = (const float*)d_in[5];
  const float* patch_w= (const float*)d_in[6];
  const float* patch_b= (const float*)d_in[7];
  const float* pos_z  = (const float*)d_in[8];
  const float* pos_x  = (const float*)d_in[9];
  const float* ln1_s  = (const float*)d_in[10];
  const float* ln1_b  = (const float*)d_in[11];
  const float* qkv_w  = (const float*)d_in[12];
  const float* proj_w = (const float*)d_in[13];
  const float* ln2_s  = (const float*)d_in[14];
  const float* ln2_b  = (const float*)d_in[15];
  const float* mlp1_w = (const float*)d_in[16];
  const float* mlp2_w = (const float*)d_in[17];
  const float* norm_s = (const float*)d_in[18];
  const float* norm_b = (const float*)d_in[19];
  float* out = (float*)d_out;

  char* W = (char*)d_ws;
  bf16* WqkvT  = (bf16*)(W + O_WQKV);
  bf16* WprojT = (bf16*)(W + O_WPROJ);
  bf16* Wmlp1T = (bf16*)(W + O_WMLP1);
  bf16* Wmlp2T = (bf16*)(W + O_WMLP2);
  bf16* WpatchT= (bf16*)(W + O_WPATCH);
  float* conv18 = (float*)(W + O_CONV);
  float* conv16 = conv18 + 16 * 16 * 196;
  float* conv14 = conv16 + 16 * 16 * 256;
  float* tcat = (float*)(W + O_TCAT);
  float* msm  = (float*)(W + O_MSM);
  float* wvec = (float*)(W + O_W);
  int*   oidx = (int*)(W + O_OIDX);
  float* tz   = (float*)(W + O_TZ);
  float* tx   = (float*)(W + O_TX);
  float* tc   = (float*)(W + O_TC);
  float* tsum = (float*)(W + O_TSUM);
  bf16* hbf   = (bf16*)(W + O_H);
  bf16* qkvbf = (bf16*)(W + O_QKV);
  bf16* obf   = (bf16*)(W + O_OB);
  bf16* ubf   = (bf16*)(W + O_U);

  const int T = 256;
  // weights -> bf16 transposed
  for (int i = 0; i < 4; ++i) {
    tcvt_kern<<<(384 * 1152 + T - 1) / T, T, 0, stream>>>(qkv_w + (size_t)i * 384 * 1152, WqkvT + (size_t)i * 1152 * 384, 384, 1152);
    tcvt_kern<<<(384 * 384 + T - 1) / T, T, 0, stream>>>(proj_w + (size_t)i * 384 * 384, WprojT + (size_t)i * 384 * 384, 384, 384);
    tcvt_kern<<<(384 * 1536 + T - 1) / T, T, 0, stream>>>(mlp1_w + (size_t)i * 384 * 1536, Wmlp1T + (size_t)i * 1536 * 384, 384, 1536);
    tcvt_kern<<<(1536 * 384 + T - 1) / T, T, 0, stream>>>(mlp2_w + (size_t)i * 1536 * 384, Wmlp2T + (size_t)i * 384 * 1536, 1536, 384);
  }
  cvt_kern<<<(384 * 768 + T - 1) / T, T, 0, stream>>>(patch_w, WpatchT, 384 * 768);

  // convs + pool + tokens
  conv_kern<<<(16 * 16 * 14 * 14 + T - 1) / T, T, 0, stream>>>(x, cs18_w, conv18, 18, 14);
  conv_kern<<<(16 * 16 * 16 * 16 + T - 1) / T, T, 0, stream>>>(x, cs16_w, conv16, 16, 16);
  conv_kern<<<(16 * 16 * 18 * 18 + T - 1) / T, T, 0, stream>>>(x, cs14_w, conv14, 14, 18);
  pool_kern<<<(16 * 16 * 16 + T - 1) / T, T, 0, stream>>>(conv18, tcat, 14, 4, 0);
  pool_kern<<<(16 * 16 * 25 + T - 1) / T, T, 0, stream>>>(conv16, tcat, 16, 5, 16);
  pool_kern<<<(16 * 16 * 36 + T - 1) / T, T, 0, stream>>>(conv14, tcat, 18, 6, 41);
  msm_kern<<<(16 * 77 * 16 + T - 1) / T, T, 0, stream>>>(tcat, csMP_w, msm);
  cm_kern<<<16, 256, 0, stream>>>(msm, wvec, oidx, out + OV_OFF, (int*)(out + OI_OFF));

  // raw + channel-gated copies
  cg_kern<<<(16 * 16 * 16384 + T - 1) / T, T, 0, stream>>>(z, wvec, out + Z_OFF, out + ZCG_OFF, 16384);
  cg_kern<<<(16 * 16 * 65536 + T - 1) / T, T, 0, stream>>>(x, wvec, out + X_OFF, out + XCG_OFF, 65536);

  // patch embedding (5 groups fused into M)
  patch_k<<<dim3(6, 5120 / 64), 128, 0, stream>>>(z, 128, 64, 8, oidx, WpatchT, patch_b, tz);
  patch_k<<<dim3(6, 20480 / 64), 128, 0, stream>>>(x, 256, 256, 16, oidx, WpatchT, patch_b, tx);

  Ctx c{WqkvT, WprojT, Wmlp1T, Wmlp2T, hbf, qkvbf, obf, ubf, ln1_s, ln1_b, ln2_s, ln2_b, stream};
  for (int i = 0; i < 2; ++i) {
    run_block(c, tz, 5120, 64, i);
    run_block(c, tx, 20480, 256, i);
  }
  cat_kern<<<(80 * 320 * 384 + T - 1) / T, T, 0, stream>>>(tz, tx, pos_z, pos_x, tc);
  for (int i = 2; i < 4; ++i) run_block(c, tc, 25600, 320, i);
  sum5_kern<<<(16 * 320 * 384 + T - 1) / T, T, 0, stream>>>(tc, tsum);
  ln_k<0><<<(16 * 320) / 8, 256, 0, stream>>>(tsum, norm_s, norm_b, out + OUT0);
}
